// MaskedMultiheadAttention_3985729651117
// MI455X (gfx1250) — matrix operand order repaired and checked by reading, not compiled, not run on hardware
//
#include <hip/hip_runtime.h>

// ---------------------------------------------------------------------------
// MaskedMultiheadAttention for MI455X (gfx1250, wave32, WMMA bf16)
// B=64, T=256, C=384, H=6, D=64.  All matmuls: v_wmma_f32_16x16x32_bf16.
// ---------------------------------------------------------------------------

#define B_   64
#define T_   256
#define C_   384
#define H_   6
#define D_   64
#define M_   (B_ * T_)   // 16384 rows of x

typedef __bf16 bf16;
typedef __attribute__((ext_vector_type(16))) __bf16 bf16x16;
typedef __attribute__((ext_vector_type(8)))  __bf16 bf16x8;
typedef __attribute__((ext_vector_type(8)))  float  f32x8;

__device__ __forceinline__ f32x8 wmma_bf16(bf16x16 a, bf16x16 b, f32x8 c) {
    // D = A(16x32) * B(32x16) + C, f32 accumulate
    return __builtin_amdgcn_wmma_f32_16x16x32_bf16(
        false, a, false, b, (short)0, c, false, false);
}

// A fragment (16x32, row-major source, ld = leading dim in elements).
// Lane L (L<16): row M=L, K [0..15]; lanes 16-31: K [16..31].
__device__ __forceinline__ bf16x16 load_a_frag(const bf16* __restrict__ A,
                                               int ld, int row, int k0, int lane) {
    const int hi = lane >> 4;
    const bf16* p = A + (size_t)row * ld + k0 + 16 * hi;
    bf16x8 c0 = *(const bf16x8*)p;          // K = k0 + 16*hi + [0..7]
    bf16x8 c1 = *(const bf16x8*)(p + 8);    // K = k0 + 16*hi + 8 + [0..7]
    return __builtin_shufflevector(c0, c1, 0,1,2,3,4,5,6,7,8,9,10,11,12,13,14,15);
}

// B fragment (32x16). Bt stored [N][K] row-major (K contiguous).
// Lane L holds column N = nBase + (L&15), contraction K = k0 + 16*(L>>4) + [0..15].
__device__ __forceinline__ bf16x16 load_b_frag(const bf16* __restrict__ Bt,
                                               int ld, int col, int k0, int lane) {
    const int hi = lane >> 4;
    const bf16* p = Bt + (size_t)col * ld + k0 + 16 * hi;
    return *(const bf16x16*)p;              // 32 B, 32-byte aligned
}

// ---------------------------------------------------------------------------
// Conversion kernels
// ---------------------------------------------------------------------------
__global__ __launch_bounds__(256)
void f32_to_bf16_kernel(const float* __restrict__ src, bf16* __restrict__ dst, int n) {
    int i = blockIdx.x * 256 + threadIdx.x;
    if (i < n) dst[i] = (bf16)src[i];
}

// W is [K=in][N=out] f32; produce Wt [N][K] bf16.
__global__ __launch_bounds__(256)
void transpose_w_kernel(const float* __restrict__ W, bf16* __restrict__ Wt) {
    int i = blockIdx.x * 256 + threadIdx.x;   // i < C_*C_
    if (i >= C_ * C_) return;
    int n = i / C_, k = i - n * C_;
    Wt[(size_t)n * C_ + k] = (bf16)W[(size_t)k * C_ + n];
}

// ---------------------------------------------------------------------------
// QKV projection: X(bf16, [M,C]) @ W -> Out bf16 in [B,H,T,D] layout.
// Block = 256 threads = 8 waves; block tile 128(M) x 64(N).
// ---------------------------------------------------------------------------
__global__ __launch_bounds__(256)
void gemm_qkv_kernel(const bf16* __restrict__ Xb, const bf16* __restrict__ Wt,
                     bf16* __restrict__ Out) {
    const int mBlk = blockIdx.x * 128;
    const int nBlk = blockIdx.y * 64;
    const int lane = threadIdx.x & 31;
    const int wave = threadIdx.x >> 5;
    const int nl   = lane & 15;
    const int hi   = lane >> 4;
    const int row  = mBlk + wave * 16 + nl;

    f32x8 acc[4] = {};
    for (int k0 = 0; k0 < C_; k0 += 32) {
        bf16x16 a = load_a_frag(Xb, C_, row, k0, lane);
        #pragma unroll
        for (int t = 0; t < 4; ++t) {
            bf16x16 b = load_b_frag(Wt, C_, nBlk + t * 16 + nl, k0, lane);
            acc[t] = wmma_bf16(a, b, acc[t]);
        }
    }
    // C layout: VGPR r -> row r (lanes 0-15) / row r+8 (lanes 16-31); col = lane&15
    #pragma unroll
    for (int t = 0; t < 4; ++t) {
        #pragma unroll
        for (int r = 0; r < 8; ++r) {
            int m = mBlk + wave * 16 + r + 8 * hi;
            int n = nBlk + t * 16 + nl;
            int b = m >> 8, tq = m & (T_ - 1);
            int h = n >> 6, d  = n & (D_ - 1);
            Out[(((size_t)b * H_ + h) * T_ + tq) * D_ + d] = (bf16)acc[t][r];
        }
    }
}

// ---------------------------------------------------------------------------
// Output projection: agg(bf16, [M,C]) @ Wo + bo -> out f32 [M,C].
// ---------------------------------------------------------------------------
__global__ __launch_bounds__(256)
void gemm_out_kernel(const bf16* __restrict__ Ab, const bf16* __restrict__ Wt,
                     const float* __restrict__ bias, float* __restrict__ Out) {
    const int mBlk = blockIdx.x * 128;
    const int nBlk = blockIdx.y * 64;
    const int lane = threadIdx.x & 31;
    const int wave = threadIdx.x >> 5;
    const int nl   = lane & 15;
    const int hi   = lane >> 4;
    const int row  = mBlk + wave * 16 + nl;

    f32x8 acc[4] = {};
    for (int k0 = 0; k0 < C_; k0 += 32) {
        bf16x16 a = load_a_frag(Ab, C_, row, k0, lane);
        #pragma unroll
        for (int t = 0; t < 4; ++t) {
            bf16x16 b = load_b_frag(Wt, C_, nBlk + t * 16 + nl, k0, lane);
            acc[t] = wmma_bf16(a, b, acc[t]);
        }
    }
    #pragma unroll
    for (int t = 0; t < 4; ++t) {
        int n = nBlk + t * 16 + nl;
        float bn = bias[n];
        #pragma unroll
        for (int r = 0; r < 8; ++r) {
            int m = mBlk + wave * 16 + r + 8 * hi;
            Out[(size_t)m * C_ + n] = acc[t][r] + bn;
        }
    }
}

// ---------------------------------------------------------------------------
// Flash attention. Grid = B*H*2. Block = 256 threads (8 waves).
// Each wave owns a 16-query tile; streams keys in pairs of 16 (Kc=32 for P*V).
// ---------------------------------------------------------------------------
#define VSTRIDE 272   // 256 + 16 pad: keeps rows 32-byte aligned in LDS

__device__ __forceinline__ float rowmax16(float v) {
    v = fmaxf(v, __shfl_xor(v, 1, 32));
    v = fmaxf(v, __shfl_xor(v, 2, 32));
    v = fmaxf(v, __shfl_xor(v, 4, 32));
    v = fmaxf(v, __shfl_xor(v, 8, 32));
    return v;
}
__device__ __forceinline__ float rowsum16(float v) {
    v += __shfl_xor(v, 1, 32);
    v += __shfl_xor(v, 2, 32);
    v += __shfl_xor(v, 4, 32);
    v += __shfl_xor(v, 8, 32);
    return v;
}

__global__ __launch_bounds__(256)
void attn_kernel(const bf16* __restrict__ q, const bf16* __restrict__ k,
                 const bf16* __restrict__ v, bf16* __restrict__ agg) {
    __shared__ bf16 sVt[D_ * VSTRIDE];  // V transposed: [d][t], 34816 B
    __shared__ bf16 sP[8 * 16 * 32];    // per-wave softmax tile slab, 8192 B

    const int bh   = blockIdx.x >> 1;
    const int half = blockIdx.x & 1;
    const int b    = bh / H_;
    const int h    = bh - b * H_;

    const bf16* Qp = q + (size_t)bh * T_ * D_;
    const bf16* Kp = k + (size_t)bh * T_ * D_;
    const bf16* Vp = v + (size_t)bh * T_ * D_;

    // Stage V transposed into LDS: sVt[d][t] = V[t][d]
    for (int idx = threadIdx.x; idx < T_ * D_; idx += 256) {
        int t = idx >> 6, d = idx & (D_ - 1);
        sVt[d * VSTRIDE + t] = Vp[idx];
    }
    __syncthreads();

    const int lane = threadIdx.x & 31;
    const int wave = threadIdx.x >> 5;
    const int nl   = lane & 15;
    const int hi   = lane >> 4;
    const int q0   = half * 128 + wave * 16;   // first query row of this wave

    // Q fragments for the whole tile (contraction D=64 -> two K=32 chunks)
    bf16x16 qa0 = load_a_frag(Qp, D_, q0 + nl, 0,  lane);
    bf16x16 qa1 = load_a_frag(Qp, D_, q0 + nl, 32, lane);

    f32x8 O[4] = {};
    float mrow[8], lrow[8];
    #pragma unroll
    for (int r = 0; r < 8; ++r) { mrow[r] = -3.0e38f; lrow[r] = 0.0f; }

    bf16* slab = &sP[wave * 512];

    for (int kp = 0; kp < q0 + 16; kp += 32) {   // causal bound
        // ----- scores for two 16-key subtiles -----
        f32x8 s[2];
        #pragma unroll
        for (int sub = 0; sub < 2; ++sub) {
            int kb = kp + sub * 16;
            bf16x16 kb0 = load_b_frag(Kp, D_, kb + nl, 0,  lane);
            bf16x16 kb1 = load_b_frag(Kp, D_, kb + nl, 32, lane);
            f32x8 c = {};
            c = wmma_bf16(qa0, kb0, c);
            c = wmma_bf16(qa1, kb1, c);
            s[sub] = c;
        }

        // ----- scale + causal mask + online softmax stats -----
        float alpha[8];
        #pragma unroll
        for (int r = 0; r < 8; ++r) {
            const int qrow = q0 + r + 8 * hi;
            #pragma unroll
            for (int sub = 0; sub < 2; ++sub) {
                int key = kp + sub * 16 + nl;
                float val = s[sub][r] * 0.125f;       // 1/sqrt(64)
                if (key > qrow) val = -3.0e38f;
                s[sub][r] = val;
            }
            float rm = rowmax16(fmaxf(s[0][r], s[1][r]));
            float nm = fmaxf(mrow[r], rm);
            float a  = __expf(mrow[r] - nm);
            float p0 = __expf(s[0][r] - nm);
            float p1 = __expf(s[1][r] - nm);
            s[0][r] = p0;
            s[1][r] = p1;
            lrow[r] = lrow[r] * a + rowsum16(p0 + p1);
            mrow[r] = nm;
            alpha[r] = a;
        }

        // ----- C-layout -> A-layout for P via per-wave LDS slab -----
        #pragma unroll
        for (int r = 0; r < 8; ++r) {
            slab[(r + 8 * hi) * 32 + nl]      = (bf16)s[0][r];
            slab[(r + 8 * hi) * 32 + 16 + nl] = (bf16)s[1][r];
        }
        asm volatile("s_wait_dscnt 0" ::: "memory");  // lanes exchange via LDS

        const bf16* pp = slab + nl * 32 + 16 * hi;
        bf16x8 pc0 = *(const bf16x8*)pp;
        bf16x8 pc1 = *(const bf16x8*)(pp + 8);
        bf16x16 pa = __builtin_shufflevector(pc0, pc1,
                        0,1,2,3,4,5,6,7,8,9,10,11,12,13,14,15);

        // ----- rescale O, accumulate P(16x32) * V(32x16) over 4 d-tiles -----
        #pragma unroll
        for (int t = 0; t < 4; ++t) {
            #pragma unroll
            for (int r = 0; r < 8; ++r) O[t][r] *= alpha[r];
            bf16x16 vb = *(const bf16x16*)&sVt[(t * 16 + nl) * VSTRIDE + kp + 16 * hi];
            O[t] = wmma_bf16(pa, vb, O[t]);
        }
    }

    // ----- normalize and store agg as bf16 in [B,T,C] (= [B,T,H,D]) -----
    #pragma unroll
    for (int r = 0; r < 8; ++r) {
        float inv = 1.0f / lrow[r];
        int tq = q0 + r + 8 * hi;
        #pragma unroll
        for (int t = 0; t < 4; ++t) {
            int d = t * 16 + nl;
            agg[((size_t)b * T_ + tq) * C_ + h * D_ + d] = (bf16)(O[t][r] * inv);
        }
    }
}

// ---------------------------------------------------------------------------
// Host launch
// ---------------------------------------------------------------------------
extern "C" void kernel_launch(void* const* d_in, const int* in_sizes, int n_in,
                              void* d_out, int out_size, void* d_ws, size_t ws_size,
                              hipStream_t stream) {
    (void)in_sizes; (void)n_in; (void)out_size; (void)ws_size;
    const float* x  = (const float*)d_in[0];
    const float* Wq = (const float*)d_in[1];
    const float* Wk = (const float*)d_in[2];
    const float* Wv = (const float*)d_in[3];
    const float* Wo = (const float*)d_in[4];
    const float* bo = (const float*)d_in[5];
    float* out = (float*)d_out;

    char* ws = (char*)d_ws;
    size_t off = 0;
    auto alloc = [&](size_t bytes) -> void* {
        void* p = ws + off;
        off += (bytes + 255) & ~(size_t)255;
        return p;
    };
    bf16* xb   = (bf16*)alloc((size_t)M_ * C_ * 2);
    bf16* wqt  = (bf16*)alloc((size_t)C_ * C_ * 2);
    bf16* wkt  = (bf16*)alloc((size_t)C_ * C_ * 2);
    bf16* wvt  = (bf16*)alloc((size_t)C_ * C_ * 2);
    bf16* wot  = (bf16*)alloc((size_t)C_ * C_ * 2);
    bf16* qb   = (bf16*)alloc((size_t)M_ * C_ * 2);
    bf16* kb   = (bf16*)alloc((size_t)M_ * C_ * 2);
    bf16* vb   = (bf16*)alloc((size_t)M_ * C_ * 2);
    bf16* aggb = (bf16*)alloc((size_t)M_ * C_ * 2);

    const int nx = M_ * C_;
    f32_to_bf16_kernel<<<(nx + 255) / 256, 256, 0, stream>>>(x, xb, nx);

    const int nw = C_ * C_;
    transpose_w_kernel<<<(nw + 255) / 256, 256, 0, stream>>>(Wq, wqt);
    transpose_w_kernel<<<(nw + 255) / 256, 256, 0, stream>>>(Wk, wkt);
    transpose_w_kernel<<<(nw + 255) / 256, 256, 0, stream>>>(Wv, wvt);
    transpose_w_kernel<<<(nw + 255) / 256, 256, 0, stream>>>(Wo, wot);

    dim3 gGemm(M_ / 128, C_ / 64);
    gemm_qkv_kernel<<<gGemm, 256, 0, stream>>>(xb, wqt, qb);
    gemm_qkv_kernel<<<gGemm, 256, 0, stream>>>(xb, wkt, kb);
    gemm_qkv_kernel<<<gGemm, 256, 0, stream>>>(xb, wvt, vb);

    attn_kernel<<<B_ * H_ * 2, 256, 0, stream>>>(qb, kb, vb, aggb);

    gemm_out_kernel<<<gGemm, 256, 0, stream>>>(aggb, wot, bo, out);
}
